// GCNv2_LCG_15839839387880
// MI455X (gfx1250) — matrix operand order repaired and checked by reading, not compiled, not run on hardware
//
#include <hip/hip_runtime.h>
#include <hip/hip_bf16.h>
#include <stdint.h>

#define DIM 128

typedef __bf16 v16bf __attribute__((ext_vector_type(16)));
typedef float  v8f   __attribute__((ext_vector_type(8)));
typedef unsigned short u16;

union AFrag { unsigned u[8]; v16bf v; };
union CFrag { float f[8]; v8f v; };

static __device__ __forceinline__ u16 f2bf_rne(float f) {
  unsigned u = __float_as_uint(f);
  u += 0x7FFFu + ((u >> 16) & 1u);   // round-to-nearest-even
  return (u16)(u >> 16);
}

// ---------------------------------------------------------------------------
// Pack f32 weight [K][128] into bf16 WMMA B-fragment order:
// elem d -> j=d&15, lane=(d>>4)&31, rest=d>>9, kc=rest%Kc, nt=rest/Kc
// k = kc*32 + (j>>3)*16 + (lane>>4)*8 + (j&7) ; n = nt*16 + (lane&15)
// Each lane's 16 bf16 are contiguous => one 32B vector load in the GEMM.
// ---------------------------------------------------------------------------
__global__ void pack_w_kernel(const float* __restrict__ src, u16* __restrict__ dst,
                              int K, int total) {
  int d = blockIdx.x * blockDim.x + threadIdx.x;
  if (d >= total) return;
  int j    = d & 15;
  int lane = (d >> 4) & 31;
  int rest = d >> 9;
  int Kc   = K >> 5;
  int kc   = rest % Kc;
  int nt   = rest / Kc;
  int k    = kc * 32 + ((j >> 3) << 4) + ((lane >> 4) << 3) + (j & 7);
  int n    = (nt << 4) + (lane & 15);
  dst[d] = f2bf_rne(src[(size_t)k * DIM + n]);
}

// ---------------------------------------------------------------------------
// Degrees + inverse norm per edge
// ---------------------------------------------------------------------------
__global__ void deg_kernel(const int* __restrict__ le, const int* __restrict__ ce,
                           float* __restrict__ ldeg, float* __restrict__ cdeg, int E) {
  int e = blockIdx.x * blockDim.x + threadIdx.x;
  if (e >= E) return;
  unsafeAtomicAdd(&ldeg[le[e]], 1.0f);
  unsafeAtomicAdd(&cdeg[ce[e]], 1.0f);
}

__global__ void inv_kernel(const int* __restrict__ le, const int* __restrict__ ce,
                           const float* __restrict__ ldeg, const float* __restrict__ cdeg,
                           float* __restrict__ inv, int E) {
  int e = blockIdx.x * blockDim.x + threadIdx.x;
  if (e >= E) return;
  inv[e] = rsqrtf(ldeg[le[e]] * cdeg[ce[e]]);
}

// ---------------------------------------------------------------------------
// Edge scatter (both directions, float4 per thread, hw f32 atomics)
// l2c_aggr[ce[e]] += l2c_msg[le[e]] * inv[e]
// c2l_aggr[le[e]] += c2l_msg[ce[e]] * inv[e]
// ---------------------------------------------------------------------------
__global__ void scatter_kernel(const float* __restrict__ l2c_msg,
                               const float* __restrict__ c2l_msg,
                               const float* __restrict__ inv,
                               const int* __restrict__ le, const int* __restrict__ ce,
                               float* __restrict__ l2c_aggr, float* __restrict__ c2l_aggr,
                               int E) {
  int idx = blockIdx.x * blockDim.x + threadIdx.x;
  int total = E << 5;                 // E * 32 quads
  if (idx >= total) return;
  int e = idx >> 5;
  int j = (idx & 31) << 2;
  float w  = inv[e];
  int li = le[e], ci = ce[e];
  float4 a = *(const float4*)(l2c_msg + (size_t)li * DIM + j);
  float* d0 = l2c_aggr + (size_t)ci * DIM + j;
  unsafeAtomicAdd(d0 + 0, a.x * w);
  unsafeAtomicAdd(d0 + 1, a.y * w);
  unsafeAtomicAdd(d0 + 2, a.z * w);
  unsafeAtomicAdd(d0 + 3, a.w * w);
  float4 b = *(const float4*)(c2l_msg + (size_t)ci * DIM + j);
  float* d1 = c2l_aggr + (size_t)li * DIM + j;
  unsafeAtomicAdd(d1 + 0, b.x * w);
  unsafeAtomicAdd(d1 + 1, b.y * w);
  unsafeAtomicAdd(d1 + 2, b.z * w);
  unsafeAtomicAdd(d1 + 3, b.w * w);
}

// ---------------------------------------------------------------------------
// A-fragment build from LDS bf16 rows (ISA 7.12.2 16-bit A 16x32 layout):
// VGPR v, half=lane>>4 : K pair base = ((v&4)?16:0) + half*8 + (v&3)*2
// ---------------------------------------------------------------------------
template <int KC>
static __device__ __forceinline__ void load_afrags(const u16* s, int lr, int K,
                                                   int half, AFrag* A) {
#pragma unroll
  for (int kc = 0; kc < KC; ++kc) {
#pragma unroll
    for (int v = 0; v < 8; ++v) {
      int k = kc * 32 + ((v & 4) ? 16 : 0) + (half << 3) + ((v & 3) << 1);
      A[kc].u[v] = *(const unsigned*)(s + (size_t)lr * K + k);
    }
  }
}

// ---------------------------------------------------------------------------
// Fused 2-layer MLP: Y = relu(X@W1+b1)@W2+b2, X:[N,128] f32, optional
// pair-swap on input rows (row ^ 1). 4 waves/block, 64 rows/block.
// ---------------------------------------------------------------------------
__global__ __launch_bounds__(128) void mlp2_kernel(
    const float* __restrict__ X, int N, int swap,
    const u16* __restrict__ pW1, const float* __restrict__ b1,
    const u16* __restrict__ pW2, const float* __restrict__ b2,
    float* __restrict__ Y) {
  __shared__ u16 sX[64 * DIM];
  __shared__ u16 sH[64 * DIM];
  const int tid  = threadIdx.x;
  const int row0 = blockIdx.x * 64;

  // stage X tile to LDS as bf16
  for (int i = tid; i < 64 * 32; i += 128) {
    int r = i >> 5, c4 = (i & 31) << 2;
    int gr = row0 + r;
    float4 val = make_float4(0.f, 0.f, 0.f, 0.f);
    if (gr < N) {
      int sr = swap ? (gr ^ 1) : gr;
      val = *(const float4*)(X + (size_t)sr * DIM + c4);
    }
    unsigned lo = ((unsigned)f2bf_rne(val.y) << 16) | f2bf_rne(val.x);
    unsigned hi = ((unsigned)f2bf_rne(val.w) << 16) | f2bf_rne(val.z);
    *(uint2*)&sX[r * DIM + c4] = make_uint2(lo, hi);
  }
  __syncthreads();

  const int lane = tid & 31;
  const int half = lane >> 4;
  const int col  = lane & 15;
  const int wrow = (tid >> 5) << 4;   // wave's 16-row tile base (local)
  const int lr   = wrow + col;        // this lane's A row (local)

  AFrag A[4];
  load_afrags<4>(sX, lr, DIM, half, A);

  // layer 1: h = relu(x@W1+b1), restage to LDS in A-layout order
#pragma unroll
  for (int nt = 0; nt < 8; ++nt) {
    float bv = b1[(nt << 4) + col];
    CFrag c;
#pragma unroll
    for (int i = 0; i < 8; ++i) c.f[i] = bv;
#pragma unroll
    for (int kc = 0; kc < 4; ++kc) {
      v16bf B = *(const v16bf*)(pW1 + ((size_t)(((nt << 2) + kc) * 32 + lane) << 4));
      c.v = __builtin_amdgcn_wmma_f32_16x16x32_bf16(false, A[kc].v, false, B,
                                                    (short)0, c.v, false, false);
    }
#pragma unroll
    for (int v = 0; v < 8; ++v) {
      float h = c.f[v] > 0.f ? c.f[v] : 0.f;
      sH[(wrow + v + (half << 3)) * DIM + (nt << 4) + col] = f2bf_rne(h);
    }
  }
  __syncthreads();

  AFrag A2[4];
  load_afrags<4>(sH, lr, DIM, half, A2);

  // layer 2: y = h@W2+b2
#pragma unroll
  for (int nt = 0; nt < 8; ++nt) {
    float bv = b2[(nt << 4) + col];
    CFrag c;
#pragma unroll
    for (int i = 0; i < 8; ++i) c.f[i] = bv;
#pragma unroll
    for (int kc = 0; kc < 4; ++kc) {
      v16bf B = *(const v16bf*)(pW2 + ((size_t)(((nt << 2) + kc) * 32 + lane) << 4));
      c.v = __builtin_amdgcn_wmma_f32_16x16x32_bf16(false, A2[kc].v, false, B,
                                                    (short)0, c.v, false, false);
    }
#pragma unroll
    for (int v = 0; v < 8; ++v) {
      int grow = row0 + wrow + v + (half << 3);
      if (grow < N) Y[(size_t)grow * DIM + (nt << 4) + col] = c.f[v];
    }
  }
}

// ---------------------------------------------------------------------------
// Update linear on concat of P parts: Y = [X0|X1|X2]@W + b, K = 128*P
// ---------------------------------------------------------------------------
template <int P>
__global__ __launch_bounds__(128) void update_kernel(
    const float* __restrict__ X0, const float* __restrict__ X1,
    const float* __restrict__ X2, int N,
    const u16* __restrict__ pW, const float* __restrict__ bias,
    float* __restrict__ Y) {
  constexpr int K  = DIM * P;
  constexpr int KC = K / 32;
  __shared__ u16 sX[64 * K];
  const int tid  = threadIdx.x;
  const int row0 = blockIdx.x * 64;

  for (int i = tid; i < 64 * 32 * P; i += 128) {
    int p   = i / (64 * 32);
    int rem = i - p * (64 * 32);
    int r = rem >> 5, c4 = (rem & 31) << 2;
    const float* Xp = (p == 0) ? X0 : ((p == 1) ? X1 : X2);
    int gr = row0 + r;
    float4 val = make_float4(0.f, 0.f, 0.f, 0.f);
    if (gr < N) val = *(const float4*)(Xp + (size_t)gr * DIM + c4);
    unsigned lo = ((unsigned)f2bf_rne(val.y) << 16) | f2bf_rne(val.x);
    unsigned hi = ((unsigned)f2bf_rne(val.w) << 16) | f2bf_rne(val.z);
    *(uint2*)&sX[r * K + p * DIM + c4] = make_uint2(lo, hi);
  }
  __syncthreads();

  const int lane = tid & 31;
  const int half = lane >> 4;
  const int col  = lane & 15;
  const int wrow = (tid >> 5) << 4;
  const int lr   = wrow + col;

  AFrag A[KC];
  load_afrags<KC>(sX, lr, K, half, A);

#pragma unroll
  for (int nt = 0; nt < 8; ++nt) {
    float bv = bias[(nt << 4) + col];
    CFrag c;
#pragma unroll
    for (int i = 0; i < 8; ++i) c.f[i] = bv;
#pragma unroll
    for (int kc = 0; kc < KC; ++kc) {
      v16bf B = *(const v16bf*)(pW + ((size_t)((nt * KC + kc) * 32 + lane) << 4));
      c.v = __builtin_amdgcn_wmma_f32_16x16x32_bf16(false, A[kc].v, false, B,
                                                    (short)0, c.v, false, false);
    }
#pragma unroll
    for (int v = 0; v < 8; ++v) {
      int grow = row0 + wrow + v + (half << 3);
      if (grow < N) Y[(size_t)grow * DIM + (nt << 4) + col] = c.f[v];
    }
  }
}

// ---------------------------------------------------------------------------
// Host orchestration
// ---------------------------------------------------------------------------
extern "C" void kernel_launch(void* const* d_in, const int* in_sizes, int n_in,
                              void* d_out, int out_size, void* d_ws, size_t ws_size,
                              hipStream_t stream) {
  const int* le = (const int*)d_in[2];
  const int* ce = (const int*)d_in[3];
  const float* l_emb = (const float*)d_in[4];
  const float* c_emb = (const float*)d_in[5];

  const int E = in_sizes[2];
  const int L = in_sizes[4] / DIM;
  const int C = in_sizes[5] / DIM;
  const int T = in_sizes[6] / (DIM * DIM);

  const float* W_src[8] = {(const float*)d_in[6],  (const float*)d_in[8],
                           (const float*)d_in[10], (const float*)d_in[12],
                           (const float*)d_in[14], (const float*)d_in[16],
                           (const float*)d_in[18], (const float*)d_in[20]};
  const int   W_K[8]   = {128, 128, 128, 128, 128, 128, 256, 384};
  const float* l2c_b1 = (const float*)d_in[7],  *l2c_b2 = (const float*)d_in[9];
  const float* c2l_b1 = (const float*)d_in[11], *c2l_b2 = (const float*)d_in[13];
  const float* l2l_b1 = (const float*)d_in[15], *l2l_b2 = (const float*)d_in[17];
  const float* cup_b  = (const float*)d_in[19], *lup_b  = (const float*)d_in[21];

  float* out   = (float*)d_out;
  float* out_l = out;
  float* out_c = out + (size_t)(T + 1) * L * DIM;

  // workspace carve (256B aligned)
  char* w = (char*)d_ws;
  size_t off = 0;
  auto carve = [&](size_t bytes) -> void* {
    void* p = w + off;
    off = (off + bytes + 255) & ~(size_t)255;
    return p;
  };
  float* l2c_msg  = (float*)carve((size_t)L * DIM * 4);
  float* c2l_msg  = (float*)carve((size_t)C * DIM * 4);
  float* l2l_msg  = (float*)carve((size_t)L * DIM * 4);
  float* l2c_aggr = (float*)carve((size_t)C * DIM * 4);
  float* c2l_aggr = (float*)carve((size_t)L * DIM * 4);
  float* l_deg    = (float*)carve((size_t)L * 4);
  float* c_deg    = (float*)carve((size_t)C * 4);
  float* inv      = (float*)carve((size_t)E * 4);
  u16* Wp[8];
  for (int m = 0; m < 8; ++m)
    Wp[m] = (u16*)carve((size_t)T * W_K[m] * DIM * sizeof(u16));

  // pack all weights -> bf16 B-fragment layout
  for (int m = 0; m < 8; ++m) {
    int total = W_K[m] * DIM;
    for (int t = 0; t < T; ++t) {
      pack_w_kernel<<<(total + 255) / 256, 256, 0, stream>>>(
          W_src[m] + (size_t)t * total, Wp[m] + (size_t)t * total, W_K[m], total);
    }
  }

  // initial embeddings -> slice 0
  hipMemcpyAsync(out_l, l_emb, (size_t)L * DIM * 4, hipMemcpyDeviceToDevice, stream);
  hipMemcpyAsync(out_c, c_emb, (size_t)C * DIM * 4, hipMemcpyDeviceToDevice, stream);

  // degrees and edge normalization
  hipMemsetAsync(l_deg, 0, (size_t)L * 4, stream);
  hipMemsetAsync(c_deg, 0, (size_t)C * 4, stream);
  deg_kernel<<<(E + 255) / 256, 256, 0, stream>>>(le, ce, l_deg, c_deg, E);
  inv_kernel<<<(E + 255) / 256, 256, 0, stream>>>(le, ce, l_deg, c_deg, inv, E);

  const size_t PK1 = (size_t)128 * DIM, PK2 = (size_t)256 * DIM, PK3 = (size_t)384 * DIM;
  const int gL = (L + 63) / 64, gC = (C + 63) / 64;

  for (int t = 0; t < T; ++t) {
    const float* lc = out_l + (size_t)t * L * DIM;
    const float* cc = out_c + (size_t)t * C * DIM;
    float* ln = out_l + (size_t)(t + 1) * L * DIM;
    float* cn = out_c + (size_t)(t + 1) * C * DIM;

    mlp2_kernel<<<gL, 128, 0, stream>>>(lc, L, 0,
        Wp[0] + t * PK1, l2c_b1 + (size_t)t * DIM,
        Wp[1] + t * PK1, l2c_b2 + (size_t)t * DIM, l2c_msg);
    mlp2_kernel<<<gC, 128, 0, stream>>>(cc, C, 0,
        Wp[2] + t * PK1, c2l_b1 + (size_t)t * DIM,
        Wp[3] + t * PK1, c2l_b2 + (size_t)t * DIM, c2l_msg);
    mlp2_kernel<<<gL, 128, 0, stream>>>(lc, L, 1,
        Wp[4] + t * PK1, l2l_b1 + (size_t)t * DIM,
        Wp[5] + t * PK1, l2l_b2 + (size_t)t * DIM, l2l_msg);

    hipMemsetAsync(l2c_aggr, 0, (size_t)C * DIM * 4, stream);
    hipMemsetAsync(c2l_aggr, 0, (size_t)L * DIM * 4, stream);
    scatter_kernel<<<((E << 5) + 255) / 256, 256, 0, stream>>>(
        l2c_msg, c2l_msg, inv, le, ce, l2c_aggr, c2l_aggr, E);

    update_kernel<2><<<gC, 128, 0, stream>>>(cc, l2c_aggr, nullptr, C,
        Wp[6] + t * PK2, cup_b + (size_t)t * DIM, cn);
    update_kernel<3><<<gL, 128, 0, stream>>>(lc, c2l_aggr, l2l_msg, L,
        Wp[7] + t * PK3, lup_b + (size_t)t * DIM, ln);
  }
}
